// GNNLinkPredictor_70093866270944
// MI455X (gfx1250) — matrix operand order repaired and checked by reading, not compiled, not run on hardware
//
#include <hip/hip_runtime.h>

// ---------------------------------------------------------------------------
// GNN link predictor (2x GATConv + edge MLP) for gfx1250.
// Matrix phases use v_wmma_f32_16x16x32_bf16 (wave32 WMMA), activations f32.
// edge_index is cast per harness convention: integer -> const int*.
// ---------------------------------------------------------------------------

typedef __attribute__((ext_vector_type(16))) __bf16 v16bf;
typedef __attribute__((ext_vector_type(8)))  __bf16 v8bf;
typedef __attribute__((ext_vector_type(8)))  float  v8f;

#define LEAK 0.2f

__device__ __forceinline__ unsigned short f2bf_rne(float f) {
  unsigned int u = __float_as_uint(f);
  u += 0x7FFFu + ((u >> 16) & 1u);          // round-to-nearest-even
  return (unsigned short)(u >> 16);
}

__device__ __forceinline__ void atomicMaxF(float* addr, float val) {
  if (val >= 0.0f) atomicMax((int*)addr, __float_as_int(val));
  else             atomicMin((unsigned int*)addr, __float_as_uint(val));
}

// ---- elementwise f32 -> bf16 ------------------------------------------------
__global__ void k_f2bf(const float* __restrict__ in, unsigned short* __restrict__ out,
                       long long n) {
  long long i = (long long)blockIdx.x * blockDim.x + threadIdx.x;
  if (i < n) out[i] = f2bf_rne(in[i]);
}

__global__ void k_fill(float* __restrict__ p, float v, long long n) {
  long long i = (long long)blockIdx.x * blockDim.x + threadIdx.x;
  if (i < n) p[i] = v;
}

// ---- repack W [K,Nc] f32 row-major into WMMA B-fragment order --------------
// B fragment as packed here (16x16x32 bf16, contiguous k, same order as A): lane l holds column n=l%16, K-range
// koff=(l/16)*16 .. +15, 2 bf16 per VGPR.  We store each lane's 16 values
// contiguously: Bp[((tn*Kb+kb)*32 + lane)*16 + j] = W[kb*32+koff+j][tn*16+n].
__global__ void k_repackB(const float* __restrict__ W, unsigned short* __restrict__ Bp,
                          int K, int Nc) {
  int Kb = K >> 5;
  long long total = (long long)(Nc >> 4) * Kb * 512;
  long long i = (long long)blockIdx.x * blockDim.x + threadIdx.x;
  if (i >= total) return;
  int j    = (int)(i & 15);
  int lane = (int)((i >> 4) & 31);
  long long t = i >> 9;
  int kb = (int)(t % Kb);
  int tn = (int)(t / Kb);
  int n    = lane & 15;
  int koff = (lane >> 4) * 16;
  int k = (kb << 5) + koff + j;
  Bp[i] = f2bf_rne(W[(long long)k * Nc + tn * 16 + n]);
}

// ---- C[M,Nc] = A[M,K] @ B[K,Nc]  (A bf16 row-major, B pre-packed) ----------
// One wave per 16(M) x 64(N) strip: one A fragment feeds 4 WMMAs (4 N-tiles,
// 4 accumulators), cutting redundant A traffic 4x vs one-tile-per-wave.
// A fragment: lane l holds row l%16, K runs [kb*32+(l/16)*16,+8) and [+8,+16)
// -> two 16B global loads per K-step.  Nc must be a multiple of 64.
__global__ void k_gemm_wmma(const __bf16* __restrict__ A, const __bf16* __restrict__ Bp,
                            float* __restrict__ C, int M, int K, int Nc) {
  int wave = blockIdx.x * (blockDim.x >> 5) + (threadIdx.x >> 5);
  int lane = threadIdx.x & 31;
  int nstrips = Nc >> 6;
  int strips = ((M + 15) >> 4) * nstrips;
  if (wave >= strips) return;
  int tm = wave / nstrips, ts = wave % nstrips;
  int l16 = lane & 15, hgrp = lane >> 4;
  int row = tm * 16 + l16; if (row >= M) row = M - 1;
  int kbase = hgrp * 16;
  int Kb = K >> 5;
  const __bf16* ar  = A + (long long)row * K;
  const __bf16* bp0 = Bp + ((long long)(ts * 4) * Kb * 32 + lane) * 16;
  long long bstride = (long long)Kb * 512;   // elements between N-tiles
  v8f acc[4] = {};
  for (int kb = 0; kb < Kb; ++kb) {
    v8bf a0 = *(const v8bf*)(ar + kbase);
    v8bf a1 = *(const v8bf*)(ar + kbase + 8);
    ar += 32;
    v16bf a;
#pragma unroll
    for (int q = 0; q < 8; ++q) { a[q] = a0[q]; a[q + 8] = a1[q]; }
#pragma unroll
    for (int t = 0; t < 4; ++t) {
      const __bf16* bp = bp0 + (long long)t * bstride + (long long)kb * 512;
      v8bf b0 = *(const v8bf*)bp;
      v8bf b1 = *(const v8bf*)(bp + 8);
      v16bf b;
#pragma unroll
      for (int q = 0; q < 8; ++q) { b[q] = b0[q]; b[q + 8] = b1[q]; }
      acc[t] = __builtin_amdgcn_wmma_f32_16x16x32_bf16(false, a, false, b,
                                                       (short)0, acc[t], false, false);
    }
  }
#pragma unroll
  for (int t = 0; t < 4; ++t) {
#pragma unroll
    for (int r = 0; r < 8; ++r) {       // D: lane l holds col l%16, row r+8*(l/16)
      int m = tm * 16 + r + hgrp * 8;
      if (m < M) C[(long long)m * Nc + (ts * 4 + t) * 16 + l16] = acc[t][r];
    }
  }
}

// ---- per-node attention scores: a[n,h] = <h[n,h,:], att[h,:]> --------------
__global__ void k_attn(const float* __restrict__ h, const float* __restrict__ asv,
                       const float* __restrict__ adv, float* __restrict__ os,
                       float* __restrict__ od, int N, int H, int C) {
  int i = blockIdx.x * blockDim.x + threadIdx.x;
  if (i >= N * H) return;
  int n = i / H, hh = i % H;
  const float* hp = h + (long long)n * H * C + hh * C;
  float s = 0.f, d = 0.f;
  for (int c = 0; c < C; ++c) { float v = hp[c]; s += v * asv[hh * C + c]; d += v * adv[hh * C + c]; }
  os[i] = s; od[i] = d;
}

// ---- edge softmax pass 1: segment max over dst -----------------------------
__global__ void k_edge_max(const int* __restrict__ ei, int E, int N,
                           const float* __restrict__ as, const float* __restrict__ ad,
                           float* __restrict__ m, int H) {
  long long i = (long long)blockIdx.x * blockDim.x + threadIdx.x;
  long long tot = (long long)(E + N) * H;
  if (i >= tot) return;
  int hh = (int)(i % H);
  long long e = i / H;
  int s, d;
  if (e < E) { s = ei[e]; d = ei[(long long)E + e]; } else { s = d = (int)(e - E); }
  float l = as[(long long)s * H + hh] + ad[(long long)d * H + hh];
  l = l > 0.f ? l : LEAK * l;
  atomicMaxF(&m[(long long)d * H + hh], l);
}

// ---- edge softmax pass 2: e = exp(l - m[dst]);  denom[dst] += e ------------
__global__ void k_edge_expsum(const int* __restrict__ ei, int E, int N,
                              const float* __restrict__ as, const float* __restrict__ ad,
                              const float* __restrict__ m, float* __restrict__ ebuf,
                              float* __restrict__ den, int H) {
  long long i = (long long)blockIdx.x * blockDim.x + threadIdx.x;
  long long tot = (long long)(E + N) * H;
  if (i >= tot) return;
  int hh = (int)(i % H);
  long long e = i / H;
  int s, d;
  if (e < E) { s = ei[e]; d = ei[(long long)E + e]; } else { s = d = (int)(e - E); }
  float l = as[(long long)s * H + hh] + ad[(long long)d * H + hh];
  l = l > 0.f ? l : LEAK * l;
  float ex = __expf(l - m[(long long)d * H + hh]);
  ebuf[i] = ex;
  atomicAdd(&den[(long long)d * H + hh], ex);
}

// ---- edge softmax pass 3: agg[dst] += e * h[src]   (one thread per (e,ch)) -
__global__ void k_edge_agg(const int* __restrict__ ei, int E, int N,
                           const float* __restrict__ ebuf, const float* __restrict__ h,
                           float* __restrict__ agg, int H, int C) {
  long long i = (long long)blockIdx.x * blockDim.x + threadIdx.x;
  int HC = H * C;
  long long tot = (long long)(E + N) * HC;
  if (i >= tot) return;
  int ch = (int)(i % HC);
  long long e = i / HC;
  int hh = ch / C;
  int s, d;
  if (e < E) { s = ei[e]; d = ei[(long long)E + e]; } else { s = d = (int)(e - E); }
  atomicAdd(&agg[(long long)d * HC + ch],
            ebuf[e * H + hh] * h[(long long)s * HC + ch]);
}

// ---- normalize + bias (+ optional ELU), emit bf16 --------------------------
__global__ void k_finalize(const float* __restrict__ agg, const float* __restrict__ den,
                           const float* __restrict__ bias, unsigned short* __restrict__ outbf,
                           int N, int H, int C, int do_elu) {
  long long i = (long long)blockIdx.x * blockDim.x + threadIdx.x;
  int HC = H * C;
  long long tot = (long long)N * HC;
  if (i >= tot) return;
  int ch = (int)(i % HC);
  long long n = i / HC;
  int hh = ch / C;
  float v = agg[i] / den[n * H + hh] + bias[ch];
  if (do_elu) v = v > 0.f ? v : (__expf(v) - 1.f);
  outbf[i] = f2bf_rne(v);
}

// ---- edge MLP: out[e] = relu([h[u],h[v]] @ W1 + b1) @ w2 + b2 --------------
// One wave per 16-edge tile: 16x128 A (gathered bf16), 128x64 W1 -> 16 WMMAs,
// then 64->1 linear folded into a shfl_xor cross-lane reduction.
__global__ void k_edge_mlp(const __bf16* __restrict__ hbf, const int* __restrict__ ei,
                           const __bf16* __restrict__ W1p, const float* __restrict__ b1,
                           const float* __restrict__ w2, const float* __restrict__ b2p,
                           float* __restrict__ out, int E) {
  int wave = blockIdx.x * (blockDim.x >> 5) + (threadIdx.x >> 5);
  int lane = threadIdx.x & 31;
  int tiles = (E + 15) >> 4;
  if (wave >= tiles) return;
  int l16 = lane & 15, hgrp = lane >> 4;
  long long e = (long long)wave * 16 + l16;
  if (e >= E) e = E - 1;
  long long u = ei[e];
  long long v = ei[(long long)E + e];
  const __bf16* up = hbf + u * 64;
  const __bf16* vp = hbf + v * 64;
  int kbase = hgrp * 16;
  v8f acc[4] = {};
#pragma unroll
  for (int kb = 0; kb < 4; ++kb) {
    int k0 = kb * 32;
    int kg0 = k0 + kbase, kg1 = k0 + kbase + 8;
    v8bf a0 = *(const v8bf*)((kg0 < 64) ? (up + kg0) : (vp + (kg0 - 64)));
    v8bf a1 = *(const v8bf*)((kg1 < 64) ? (up + kg1) : (vp + (kg1 - 64)));
    v16bf a;
#pragma unroll
    for (int q = 0; q < 8; ++q) { a[q] = a0[q]; a[q + 8] = a1[q]; }
#pragma unroll
    for (int t = 0; t < 4; ++t) {
      const __bf16* bp = W1p + ((long long)(t * 4 + kb) * 32 + lane) * 16;
      v8bf b0 = *(const v8bf*)bp;
      v8bf b1v = *(const v8bf*)(bp + 8);
      v16bf b;
#pragma unroll
      for (int q = 0; q < 8; ++q) { b[q] = b0[q]; b[q + 8] = b1v[q]; }
      acc[t] = __builtin_amdgcn_wmma_f32_16x16x32_bf16(false, a, false, b,
                                                       (short)0, acc[t], false, false);
    }
  }
  float p[8];
#pragma unroll
  for (int r = 0; r < 8; ++r) p[r] = 0.f;
#pragma unroll
  for (int t = 0; t < 4; ++t) {
    int n = t * 16 + l16;
    float bias = b1[n], w = w2[n];
#pragma unroll
    for (int r = 0; r < 8; ++r) {
      float hv = acc[t][r] + bias;
      hv = hv > 0.f ? hv : 0.f;
      p[r] += hv * w;
    }
  }
#pragma unroll
  for (int msk = 1; msk < 16; msk <<= 1)
#pragma unroll
    for (int r = 0; r < 8; ++r) p[r] += __shfl_xor(p[r], msk, 32);
  if (l16 == 0) {
    float bb = b2p[0];
#pragma unroll
    for (int r = 0; r < 8; ++r) {
      long long oi = (long long)wave * 16 + r + hgrp * 8;
      if (oi < E) out[oi] = p[r] + bb;
    }
  }
}

// ---------------------------------------------------------------------------
extern "C" void kernel_launch(void* const* d_in, const int* in_sizes, int n_in,
                              void* d_out, int out_size, void* d_ws, size_t ws_size,
                              hipStream_t stream) {
  (void)n_in; (void)out_size; (void)ws_size;
  const float* x    = (const float*)d_in[0];
  const int*   ei   = (const int*)d_in[1];
  const float* W1   = (const float*)d_in[2];
  const float* as1w = (const float*)d_in[3];
  const float* ad1w = (const float*)d_in[4];
  const float* b1   = (const float*)d_in[5];
  const float* W2   = (const float*)d_in[6];
  const float* as2w = (const float*)d_in[7];
  const float* ad2w = (const float*)d_in[8];
  const float* b2   = (const float*)d_in[9];
  const float* mw1  = (const float*)d_in[10];
  const float* mb1  = (const float*)d_in[11];
  const float* mw2  = (const float*)d_in[12];
  const float* mb2  = (const float*)d_in[13];
  float* out = (float*)d_out;

  const int N  = in_sizes[0] / 128;   // 50000 (divisible by 16)
  const int E  = in_sizes[1] / 2;     // 800000 (divisible by 16)
  const int H1 = 4, C = 64;
  const int F1 = H1 * C;              // 256

  char* ws = (char*)d_ws;
  size_t off = 0;
  auto take = [&](size_t bytes) -> void* {
    void* p = ws + off; off += (bytes + 255) & ~(size_t)255; return p;
  };
  unsigned short* xbf   = (unsigned short*)take((size_t)N * 128 * 2);
  unsigned short* W1p   = (unsigned short*)take((size_t)16 * 4 * 512 * 2);
  unsigned short* W2p   = (unsigned short*)take((size_t)4 * 8 * 512 * 2);
  unsigned short* MW1p  = (unsigned short*)take((size_t)4 * 4 * 512 * 2);
  float* h1lin = (float*)take((size_t)N * F1 * 4);
  float* s1    = (float*)take((size_t)N * H1 * 4);
  float* d1    = (float*)take((size_t)N * H1 * 4);
  float* m1    = (float*)take((size_t)N * H1 * 4);
  float* den1  = (float*)take((size_t)N * H1 * 4);
  float* e1    = (float*)take((size_t)(E + N) * H1 * 4);
  float* agg1  = (float*)take((size_t)N * F1 * 4);
  unsigned short* h1out = (unsigned short*)take((size_t)N * F1 * 2);
  float* h2lin = (float*)take((size_t)N * C * 4);
  float* s2    = (float*)take((size_t)N * 4);
  float* d2    = (float*)take((size_t)N * 4);
  float* m2    = (float*)take((size_t)N * 4);
  float* den2  = (float*)take((size_t)N * 4);
  float* e2    = (float*)take((size_t)(E + N) * 4);
  float* agg2  = (float*)take((size_t)N * C * 4);
  unsigned short* h2out = (unsigned short*)take((size_t)N * C * 2);

  const int TB = 256;
  auto gb = [](long long n, int tb) { return (unsigned)((n + tb - 1) / tb); };
  const float NINF = -__builtin_huge_valf();

  // precision/data-movement prep: bf16 operands + B fragments packed once
  k_f2bf   <<<gb((long long)N * 128, TB), TB, 0, stream>>>(x, xbf, (long long)N * 128);
  k_repackB<<<gb(16LL * 4 * 512, TB), TB, 0, stream>>>(W1,  W1p, 128, 256);
  k_repackB<<<gb(4LL * 8 * 512, TB), TB, 0, stream>>>(W2,  W2p, 256, 64);
  k_repackB<<<gb(4LL * 4 * 512, TB), TB, 0, stream>>>(mw1, MW1p, 128, 64);

  // ---- GAT layer 1 ----
  {
    int strips = ((N + 15) / 16) * (256 / 64);
    k_gemm_wmma<<<(strips + 3) / 4, 128, 0, stream>>>((const __bf16*)xbf, (const __bf16*)W1p,
                                                      h1lin, N, 128, 256);
  }
  k_attn<<<gb((long long)N * H1, TB), TB, 0, stream>>>(h1lin, as1w, ad1w, s1, d1, N, H1, C);
  k_fill<<<gb((long long)N * H1, TB), TB, 0, stream>>>(m1, NINF, (long long)N * H1);
  k_fill<<<gb((long long)N * H1, TB), TB, 0, stream>>>(den1, 0.f, (long long)N * H1);
  k_fill<<<gb((long long)N * F1, TB), TB, 0, stream>>>(agg1, 0.f, (long long)N * F1);
  k_edge_max   <<<gb((long long)(E + N) * H1, TB), TB, 0, stream>>>(ei, E, N, s1, d1, m1, H1);
  k_edge_expsum<<<gb((long long)(E + N) * H1, TB), TB, 0, stream>>>(ei, E, N, s1, d1, m1, e1, den1, H1);
  k_edge_agg   <<<gb((long long)(E + N) * F1, TB), TB, 0, stream>>>(ei, E, N, e1, h1lin, agg1, H1, C);
  k_finalize   <<<gb((long long)N * F1, TB), TB, 0, stream>>>(agg1, den1, b1, h1out, N, H1, C, 1);

  // ---- GAT layer 2 ----
  {
    int strips = ((N + 15) / 16) * (64 / 64);
    k_gemm_wmma<<<(strips + 3) / 4, 128, 0, stream>>>((const __bf16*)h1out, (const __bf16*)W2p,
                                                      h2lin, N, 256, 64);
  }
  k_attn<<<gb((long long)N, TB), TB, 0, stream>>>(h2lin, as2w, ad2w, s2, d2, N, 1, C);
  k_fill<<<gb((long long)N, TB), TB, 0, stream>>>(m2, NINF, (long long)N);
  k_fill<<<gb((long long)N, TB), TB, 0, stream>>>(den2, 0.f, (long long)N);
  k_fill<<<gb((long long)N * C, TB), TB, 0, stream>>>(agg2, 0.f, (long long)N * C);
  k_edge_max   <<<gb((long long)(E + N), TB), TB, 0, stream>>>(ei, E, N, s2, d2, m2, 1);
  k_edge_expsum<<<gb((long long)(E + N), TB), TB, 0, stream>>>(ei, E, N, s2, d2, m2, e2, den2, 1);
  k_edge_agg   <<<gb((long long)(E + N) * C, TB), TB, 0, stream>>>(ei, E, N, e2, h2lin, agg2, 1, C);
  k_finalize   <<<gb((long long)N * C, TB), TB, 0, stream>>>(agg2, den2, b2, h2out, N, 1, C, 0);

  // ---- edge MLP (WMMA over gathered 16-edge tiles) ----
  {
    int tiles = (E + 15) / 16;
    k_edge_mlp<<<(tiles + 3) / 4, 128, 0, stream>>>((const __bf16*)h2out, ei,
                                                    (const __bf16*)MW1p, mb1, mw2, mb2, out, E);
  }
}
